// FJSPActor_721554506435
// MI455X (gfx1250) — compile-verified, matrix operand order checked by reading, not run on hardware
//
#include <hip/hip_runtime.h>
#include <hip/hip_bf16.h>

// ---------------------------------------------------------------------------
// FJSP actor head, CDNA5 (gfx1250) WMMA implementation.
//   logits[b,0]          = noop MLP(dummy)
//   logits[b,1+j*40+m]   = MLP( relu(jp[b,j]+mp[b,m]+b1) )  via v_wmma f16
//   out = [logits flat (256*2001 f32), mask as f32 (256*2001)]
// jp/mp in workspace are row-padded (64 / 48 rows per batch) so all WMMA tile
// stores are unconditional (no exec-mask ladders).
// ---------------------------------------------------------------------------

typedef _Float16 half16 __attribute__((ext_vector_type(16)));
typedef float    f32x8  __attribute__((ext_vector_type(8)));
typedef float    f32x4  __attribute__((ext_vector_type(4)));

#define BS      256
#define N_OPS   2000
#define N_JOBS  50
#define N_MA    40
#define EE      128
#define HH      128
#define NPAIR   (N_JOBS * N_MA)   // 2000
#define NLOGIT  (NPAIR + 1)       // 2001
#define NTILES  (NPAIR / 16)      // 125 (exact)
#define JP_ROWS 64                // padded rows per batch (4 tiles)
#define MP_ROWS 48                // padded rows per batch (3 tiles)

// ---------------------------------------------------------------------------
// Kernel W: swizzle W1 (Wj, Wm halves) and W2 into f16 WMMA B-fragment layout.
// Fragment f = n_tile*4 + kstep (32 per matrix); per lane 16 contiguous halves.
// B 32x16 f16 layout mirrors A 16x32: lane%16 = column n, lane/16 = k-group;
// halves i<8 -> k = g*8+i, i>=8 -> k = 16+g*8+(i-8), plus kstep*32.
// ---------------------------------------------------------------------------
__global__ void fjsp_kW(const float* __restrict__ W1, const float* __restrict__ W2,
                        _Float16* __restrict__ wsw) {
    int idx = blockIdx.x * 256 + threadIdx.x;     // 0 .. 3*16384-1
    int mtx = idx / 16384;                        // 0=Wj, 1=Wm, 2=W2
    int r   = idx % 16384;
    int i    = r & 15;
    int lane = (r >> 4) & 31;
    int f    = r >> 9;                            // 0..31
    int ntile = f >> 2, kstep = f & 3;
    int g = lane >> 4, col = lane & 15;
    int kk = (i < 8) ? (g * 8 + i) : (16 + g * 8 + (i - 8));
    int k  = kstep * 32 + kk;
    int n  = ntile * 16 + col;
    float v;
    if (mtx == 0)      v = W1[k * HH + n];
    else if (mtx == 1) v = W1[(EE + k) * HH + n];
    else               v = W2[k * HH + n];
    wsw[idx] = (_Float16)v;
}

// ---------------------------------------------------------------------------
// Kernel Proj: one wave per (tile, batch).  tiles 0..3 -> job_proj rows,
// tiles 4..6 -> ma_proj rows.  Gathers rows (via next_op for jobs), converts
// to f16 A fragments, multiplies by swizzled Wj/Wm with v_wmma.
// B fragments are double-buffered; n-tile loop fully unrolled; stores are
// unconditional into the padded jp/mp workspace.
// ---------------------------------------------------------------------------
__global__ void fjsp_kProj(const float* __restrict__ ops_emb,
                           const float* __restrict__ ma_emb,
                           const int*   __restrict__ next_op,
                           const _Float16* __restrict__ wsw,
                           float* __restrict__ jp, float* __restrict__ mp) {
    int b    = blockIdx.y;
    int t    = blockIdx.x;          // 0..6
    int lane = threadIdx.x;         // 0..31
    int g = lane >> 4, Mr = lane & 15, col = lane & 15;
    bool isJ = (t < 4);
    int  tt  = isJ ? t : (t - 4);
    int  row = tt * 16 + Mr;

    const float* src;
    if (isJ) {
        int r2 = (row < N_JOBS) ? row : 0;
        int op = next_op[b * N_JOBS + r2];
        src = ops_emb + ((size_t)b * N_OPS + op) * EE;
    } else {
        int r2 = (row < N_MA) ? row : 0;
        src = ma_emb + ((size_t)b * N_MA + r2) * EE;
    }

    // Build 4 f16 A fragments (K = 0..127 in steps of 32)
    half16 afr[4];
#pragma unroll
    for (int s = 0; s < 4; ++s) {
        int o1 = s * 32 + g * 8;
        int o2 = o1 + 16;
        f32x4 a0 = *(const f32x4*)(src + o1);
        f32x4 a1 = *(const f32x4*)(src + o1 + 4);
        f32x4 a2 = *(const f32x4*)(src + o2);
        f32x4 a3 = *(const f32x4*)(src + o2 + 4);
        half16 h;
#pragma unroll
        for (int i = 0; i < 4; ++i) {
            h[i]      = (_Float16)a0[i];
            h[4 + i]  = (_Float16)a1[i];
            h[8 + i]  = (_Float16)a2[i];
            h[12 + i] = (_Float16)a3[i];
        }
        afr[s] = h;
    }

    const _Float16* wb = wsw + (isJ ? 0 : 16384) + (size_t)lane * 16;
    float* dst = isJ ? (jp + ((size_t)b * JP_ROWS + tt * 16) * HH)
                     : (mp + ((size_t)b * MP_ROWS + tt * 16) * HH);

    // double-buffered B fragments, n-tile loop fully unrolled
    half16 bufA[4], bufB[4];
#pragma unroll
    for (int s = 0; s < 4; ++s) bufA[s] = *(const half16*)(wb + (size_t)s * 512);

#pragma unroll
    for (int nt = 0; nt < 8; ++nt) {
        const half16* cur = (nt & 1) ? bufB : bufA;
        half16*       nxt = (nt & 1) ? bufA : bufB;
        if (nt < 7) {
#pragma unroll
            for (int s = 0; s < 4; ++s)
                nxt[s] = *(const half16*)(wb + (size_t)((nt + 1) * 4 + s) * 512);
        }
        f32x8 acc = {0.f, 0.f, 0.f, 0.f, 0.f, 0.f, 0.f, 0.f};
#pragma unroll
        for (int s = 0; s < 4; ++s) {
            acc = __builtin_amdgcn_wmma_f32_16x16x32_f16(
                false, afr[s], false, cur[s], (short)0, acc, false, false);
        }
#pragma unroll
        for (int v = 0; v < 8; ++v) {
            int rr = v + 8 * g;                      // C layout: M = v + 8*(lane/16)
            dst[rr * HH + nt * 16 + col] = acc[v];   // padded -> unconditional
        }
    }
}

// ---------------------------------------------------------------------------
// Kernel Main: one block per batch. LDS holds jp[b], mp[b], b1, b2, W3 and the
// swizzled-f16 W2. Each wave builds h1 tiles in registers (relu(jp+mp+b1) ->
// f16 A fragments), runs 32 WMMAs per tile against W2 (double-buffered LDS
// B fragments, fully unrolled), and fuses the bias2 / relu / W3 dot into the
// accumulator epilogue + 16-lane butterfly reduction.
// ---------------------------------------------------------------------------
__global__ void fjsp_kMain(const float* __restrict__ jp, const float* __restrict__ mp,
                           const _Float16* __restrict__ w2sw,
                           const float* __restrict__ b1, const float* __restrict__ b2,
                           const float* __restrict__ W3, const float* __restrict__ b3,
                           float* __restrict__ out) {
    extern __shared__ char smem_raw[];
    float*    jpL = (float*)smem_raw;            // 50*128
    float*    mpL = jpL + N_JOBS * EE;           // 40*128
    float*    b1L = mpL + N_MA * EE;             // 128
    float*    b2L = b1L + EE;                    // 128
    float*    w3L = b2L + HH;                    // 128
    _Float16* w2L = (_Float16*)(w3L + HH);       // 16384 halves (32 KB)

    int b = blockIdx.x, tid = threadIdx.x;

    const float* jpg = jp + (size_t)b * JP_ROWS * EE;   // first 50 rows valid
    for (int i = tid; i < N_JOBS * EE; i += 256) jpL[i] = jpg[i];
    const float* mpg = mp + (size_t)b * MP_ROWS * EE;   // first 40 rows valid
    for (int i = tid; i < N_MA * EE; i += 256) mpL[i] = mpg[i];
    if (tid < EE) { b1L[tid] = b1[tid]; b2L[tid] = b2[tid]; w3L[tid] = W3[tid]; }
    const uint32_t* w2s32 = (const uint32_t*)w2sw;
    uint32_t*       w2l32 = (uint32_t*)w2L;
    for (int i = tid; i < 8192; i += 256) w2l32[i] = w2s32[i];
    __syncthreads();

    int wave = tid >> 5, lane = tid & 31;
    int g = lane >> 4, Mr = lane & 15, col = lane & 15;
    float b3v = b3[0];
    const _Float16* wl = w2L + (size_t)lane * 16;       // + f*512 per fragment

    for (int t = wave; t < NTILES; t += 8) {
        // --- build h1 tile (16 pairs x 128) as 4 f16 A fragments ------------
        int p = t * 16 + Mr;                     // pair index for this lane's A row
        int jj = p / N_MA, mm = p % N_MA;
        const float* jrow = jpL + jj * EE;
        const float* mrow = mpL + mm * EE;
        half16 afr[4];
#pragma unroll
        for (int s = 0; s < 4; ++s) {
            int o1 = s * 32 + g * 8;
            int o2 = o1 + 16;
            f32x4 j0 = *(const f32x4*)(jrow + o1), j1 = *(const f32x4*)(jrow + o1 + 4);
            f32x4 j2 = *(const f32x4*)(jrow + o2), j3 = *(const f32x4*)(jrow + o2 + 4);
            f32x4 m0 = *(const f32x4*)(mrow + o1), m1 = *(const f32x4*)(mrow + o1 + 4);
            f32x4 m2 = *(const f32x4*)(mrow + o2), m3 = *(const f32x4*)(mrow + o2 + 4);
            f32x4 c0 = *(const f32x4*)(b1L + o1),  c1 = *(const f32x4*)(b1L + o1 + 4);
            f32x4 c2 = *(const f32x4*)(b1L + o2),  c3 = *(const f32x4*)(b1L + o2 + 4);
            half16 h;
#pragma unroll
            for (int i = 0; i < 4; ++i) {
                float x0 = j0[i] + m0[i] + c0[i]; x0 = x0 > 0.f ? x0 : 0.f;
                float x1 = j1[i] + m1[i] + c1[i]; x1 = x1 > 0.f ? x1 : 0.f;
                float x2 = j2[i] + m2[i] + c2[i]; x2 = x2 > 0.f ? x2 : 0.f;
                float x3 = j3[i] + m3[i] + c3[i]; x3 = x3 > 0.f ? x3 : 0.f;
                h[i]      = (_Float16)x0;
                h[4 + i]  = (_Float16)x1;
                h[8 + i]  = (_Float16)x2;
                h[12 + i] = (_Float16)x3;
            }
            afr[s] = h;
        }

        // --- h2 = relu(h1 @ W2 + b2); partial logit = h2 . W3 ---------------
        float part[8];
#pragma unroll
        for (int v = 0; v < 8; ++v) part[v] = 0.f;

        half16 bufA[4], bufB[4];
#pragma unroll
        for (int s = 0; s < 4; ++s) bufA[s] = *(const half16*)(wl + (size_t)s * 512);

#pragma unroll
        for (int nt = 0; nt < 8; ++nt) {
            const half16* cur = (nt & 1) ? bufB : bufA;
            half16*       nxt = (nt & 1) ? bufA : bufB;
            if (nt < 7) {
#pragma unroll
                for (int s = 0; s < 4; ++s)
                    nxt[s] = *(const half16*)(wl + (size_t)((nt + 1) * 4 + s) * 512);
            }
            f32x8 acc = {0.f, 0.f, 0.f, 0.f, 0.f, 0.f, 0.f, 0.f};
#pragma unroll
            for (int s = 0; s < 4; ++s) {
                acc = __builtin_amdgcn_wmma_f32_16x16x32_f16(
                    false, afr[s], false, cur[s], (short)0, acc, false, false);
            }
            float b2c = b2L[nt * 16 + col];
            float w3c = w3L[nt * 16 + col];
#pragma unroll
            for (int v = 0; v < 8; ++v) {
                float hh = acc[v] + b2c;
                hh = hh > 0.f ? hh : 0.f;
                part[v] += hh * w3c;
            }
        }

        // --- reduce over the 16 columns each lane-group covers --------------
#pragma unroll
        for (int v = 0; v < 8; ++v) {
            float pv = part[v];
            pv += __shfl_xor(pv, 1, 32);
            pv += __shfl_xor(pv, 2, 32);
            pv += __shfl_xor(pv, 4, 32);
            pv += __shfl_xor(pv, 8, 32);
            if (col == 0) {
                int prow = t * 16 + v + 8 * g;   // C layout row
                out[(size_t)b * NLOGIT + 1 + prow] = pv + b3v;
            }
        }
    }
}

// ---------------------------------------------------------------------------
// Kernel Noop: tiny dummy-path MLP, broadcast to logits[:,0].
// ---------------------------------------------------------------------------
__global__ void fjsp_kNoop(const float* __restrict__ dummy, const float* __restrict__ W1,
                           const float* __restrict__ b1, const float* __restrict__ W2,
                           const float* __restrict__ b2, const float* __restrict__ W3,
                           const float* __restrict__ b3, float* __restrict__ out) {
    __shared__ float d1[HH], d2[HH], accs[256], noopv;
    int tid = threadIdx.x;
    if (tid < HH) {
        float s = b1[tid];
        for (int k = 0; k < 2 * EE; ++k) s += dummy[k] * W1[k * HH + tid];
        d1[tid] = s > 0.f ? s : 0.f;
    }
    __syncthreads();
    if (tid < HH) {
        float s = b2[tid];
        for (int k = 0; k < HH; ++k) s += d1[k] * W2[k * HH + tid];
        d2[tid] = s > 0.f ? s : 0.f;
    }
    __syncthreads();
    accs[tid] = (tid < HH) ? d2[tid] * W3[tid] : 0.f;
    __syncthreads();
    for (int st = 128; st > 0; st >>= 1) {
        if (tid < st) accs[tid] += accs[tid + st];
        __syncthreads();
    }
    if (tid == 0) noopv = accs[0] + b3[0];
    __syncthreads();
    if (tid < BS) out[(size_t)tid * NLOGIT] = noopv;
}

// ---------------------------------------------------------------------------
// Kernel Mask: bool mask -> float, second output tensor.
// ---------------------------------------------------------------------------
__global__ void fjsp_kMask(const unsigned char* __restrict__ mask, float* __restrict__ out) {
    int i = blockIdx.x * 256 + threadIdx.x;
    if (i < BS * NLOGIT) out[(size_t)BS * NLOGIT + i] = mask[i] ? 1.0f : 0.0f;
}

// ---------------------------------------------------------------------------
extern "C" void kernel_launch(void* const* d_in, const int* in_sizes, int n_in,
                              void* d_out, int out_size, void* d_ws, size_t ws_size,
                              hipStream_t stream) {
    const float* ops_emb = (const float*)d_in[0];
    const float* ma_emb  = (const float*)d_in[1];
    const int*   next_op = (const int*)d_in[2];
    const unsigned char* mask = (const unsigned char*)d_in[3];
    const float* dummy = (const float*)d_in[4];
    const float* W1 = (const float*)d_in[5];
    const float* b1 = (const float*)d_in[6];
    const float* W2 = (const float*)d_in[7];
    const float* b2 = (const float*)d_in[8];
    const float* W3 = (const float*)d_in[9];
    const float* b3 = (const float*)d_in[10];
    float* out = (float*)d_out;

    // workspace (~14.8 MB): jp f32 (padded 64 rows) | mp f32 (padded 48 rows)
    //                       | swizzled f16 {Wj, Wm, W2}
    float*    jpw  = (float*)d_ws;
    float*    mpw  = jpw + (size_t)BS * JP_ROWS * EE;
    _Float16* wsw  = (_Float16*)(mpw + (size_t)BS * MP_ROWS * EE);
    _Float16* w2sw = wsw + 2 * 16384;

    fjsp_kW<<<192, 256, 0, stream>>>(W1, W2, wsw);

    dim3 gproj(7, BS);
    fjsp_kProj<<<gproj, 32, 0, stream>>>(ops_emb, ma_emb, next_op, wsw, jpw, mpw);

    size_t smain = (size_t)(N_JOBS * EE + N_MA * EE + EE + HH + HH) * sizeof(float)
                 + (size_t)16384 * sizeof(_Float16);   // 80384 bytes
    fjsp_kMain<<<BS, 256, smain, stream>>>(jpw, mpw, w2sw, b1, b2, W3, b3, out);

    fjsp_kNoop<<<1, 256, 0, stream>>>(dummy, W1, b1, W2, b2, W3, b3, out);

    fjsp_kMask<<<(BS * NLOGIT + 255) / 256, 256, 0, stream>>>(mask, out);
}
